// RelationalCrossAttention_51230369907055
// MI455X (gfx1250) — compile-verified, matrix operand order checked by reading, not run on hardware
//
#include <hip/hip_runtime.h>
#include <cstddef>
#include <cstdint>

typedef __attribute__((ext_vector_type(16))) __bf16 v16bf;
typedef __attribute__((ext_vector_type(8)))  __bf16 v8bf;
typedef __attribute__((ext_vector_type(8)))  float  v8f;

#define SEQ 512
#define DM  1024
#define NH  16
#define NKV 4
#define HD  64
#define BSZ 4
#define ATTN_SCALE 0.125f
#define MASK_VAL  -3.4028234663852886e38f

__device__ __forceinline__ v8f wmma_bf16(v16bf a, v16bf b, v8f c) {
  return __builtin_amdgcn_wmma_f32_16x16x32_bf16(false, a, false, b, (short)0, c, false, false);
}

// Async DMA: 16 bytes global -> LDS, tracked by ASYNCcnt (no VGPR data movement).
// LDS destination address = low 32 bits of the shared-aperture flat pointer.
__device__ __forceinline__ void async_copy_b128(void* lds_dst, const void* gsrc) {
  uint32_t lds_off = (uint32_t)(uintptr_t)lds_dst;
  uint64_t gaddr   = (uint64_t)(uintptr_t)gsrc;
  asm volatile("global_load_async_to_lds_b128 %0, %1, off"
               :: "v"(lds_off), "v"(gaddr) : "memory");
}
__device__ __forceinline__ void wait_async0() {
  asm volatile("s_wait_asynccnt 0x0" ::: "memory");
}

// Load a 16x32 bf16 WMMA fragment for this lane from an LDS row `rowPtr`
// (row must be 16-byte aligned). Per ISA 7.12.2 the lane's 16 elements are the
// contiguous K-chunks [8*half, 8*half+8) and [16+8*half, 24+8*half).
// Compiles to two ds_load_b128.
__device__ __forceinline__ v16bf frag_row(const __bf16* rowPtr, int half) {
  v8bf lo = *(const v8bf*)(rowPtr + 8 * half);
  v8bf hi = *(const v8bf*)(rowPtr + 16 + 8 * half);
  return __builtin_shufflevector(lo, hi, 0, 1, 2, 3, 4, 5, 6, 7,
                                 8, 9, 10, 11, 12, 13, 14, 15);
}

// ---------------------------------------------------------------- converts
// in: [K][N] f32 -> out: [N][K] bf16 (weights pre-transposed so GEMM B-tiles
// are row-contiguous and async-copyable)
__global__ void cvt_transpose_bf16(const float* __restrict__ in, __bf16* __restrict__ out,
                                   int K, int N) {
  int idx = blockIdx.x * blockDim.x + threadIdx.x;
  if (idx >= K * N) return;
  int k = idx / N, n = idx - k * N;            // coalesced read
  out[(size_t)n * K + k] = (__bf16)in[idx];
}

// ---------------------------------------------------------------- generic GEMM
// C[M,N] = A[M,K](f32, row-major) * Bt[N,K](bf16, pre-transposed)
// block: 128 threads = 4 waves, tile 64x64, BK=32; wave -> 32x32 (2x2 WMMA tiles)
// B tile fetched by async global->LDS DMA, overlapped with A f32->bf16 staging.
template <typename CT>
__global__ void __launch_bounds__(128)
gemm_a32_b16(const float* __restrict__ A, const __bf16* __restrict__ Bt,
             CT* __restrict__ C, int M, int N, int K) {
  __shared__ alignas(16) __bf16 sA[64][40];    // [m][k], padded
  __shared__ alignas(16) __bf16 sBt[64][40];   // [n][k], padded
  const int tid  = threadIdx.x;
  const int lane = tid & 31, wave = tid >> 5;
  const int half = lane >> 4, ml = lane & 15;
  const int m0 = blockIdx.y * 64, n0 = blockIdx.x * 64;
  const int wm = (wave >> 1) * 32, wn = (wave & 1) * 32;

  v8f acc[2][2] = {};

  for (int k0 = 0; k0 < K; k0 += 32) {
    // B tile: 64 rows x 32 bf16 (64 B) = 256 x 16 B chunks; async DMA, 2/thread
#pragma unroll
    for (int q = 0; q < 2; ++q) {
      int e = q * 128 + tid;
      int r = e >> 2, c = e & 3;               // r = n row, c = 16 B chunk
      async_copy_b128(&sBt[r][c * 8], Bt + (size_t)(n0 + r) * K + k0 + c * 8);
    }
    // A tile: 64x32 f32 -> bf16 (VALU convert overlaps the B DMA)
#pragma unroll
    for (int q = 0; q < 16; ++q) {
      int e = q * 128 + tid;
      int r = e >> 5, c = e & 31;
      sA[r][c] = (__bf16)A[(size_t)(m0 + r) * K + (k0 + c)];
    }
    if (k0 + 32 < K) {                         // prefetch next A K-tile
      int r = tid >> 5, c = tid & 31;
      __builtin_prefetch(&A[(size_t)(m0 + r) * K + (k0 + 32 + c)], 0, 0);
    }
    wait_async0();
    __syncthreads();

    v16bf a[2], b[2];
#pragma unroll
    for (int t = 0; t < 2; ++t) {
      a[t] = frag_row(&sA[wm + t * 16 + ml][0], half);
      b[t] = frag_row(&sBt[wn + t * 16 + ml][0], half);
    }
#pragma unroll
    for (int tm = 0; tm < 2; ++tm)
#pragma unroll
      for (int tn = 0; tn < 2; ++tn)
        acc[tm][tn] = wmma_bf16(a[tm], b[tn], acc[tm][tn]);
    __syncthreads();
  }

#pragma unroll
  for (int tm = 0; tm < 2; ++tm)
#pragma unroll
    for (int tn = 0; tn < 2; ++tn)
#pragma unroll
      for (int r = 0; r < 8; ++r) {
        int row = m0 + wm + tm * 16 + r + 8 * half;
        int col = n0 + wn + tn * 16 + ml;
        C[(size_t)row * N + col] = (CT)acc[tm][tn][r];
      }
}

// ---------------------------------------------------------------- RoPE
// in : [b][i][h][d] f32 (GEMM output), out: [b][h][i][d] bf16
__global__ void rope_kernel(const float* __restrict__ t, const float* __restrict__ fc,
                            const float* __restrict__ fs, __bf16* __restrict__ o, int nh) {
  int idx = blockIdx.x * blockDim.x + threadIdx.x;
  int total = BSZ * SEQ * nh * (HD / 2);
  if (idx >= total) return;
  int p = idx & 31;
  int tmp = idx >> 5;
  int h = tmp % nh; tmp /= nh;
  int i = tmp & (SEQ - 1);
  int b = tmp >> 9;
  float c = fc[i * 32 + p], s = fs[i * 32 + p];
  size_t bi = ((size_t)(b * SEQ + i) * nh + h) * HD;
  float tr = t[bi + 2 * p], ti = t[bi + 2 * p + 1];
  size_t bo = ((size_t)(b * nh + h) * SEQ + i) * HD;
  o[bo + 2 * p]     = (__bf16)(tr * c - ti * s);
  o[bo + 2 * p + 1] = (__bf16)(tr * s + ti * c);
}

// ---------------------------------------------------------------- scores (QK^T, scale, mask)
// grid (j_tiles=8, i_tiles=8, b*h=64), block 128
// Q/K tiles are straight bf16 copies -> async global->LDS DMA.
__global__ void __launch_bounds__(128)
scores_kernel(const __bf16* __restrict__ qbf, const __bf16* __restrict__ kbf,
              float* __restrict__ attn) {
  __shared__ alignas(16) __bf16 sQ[64][72];
  __shared__ alignas(16) __bf16 sK[64][72];
  const int tid  = threadIdx.x;
  const int lane = tid & 31, wave = tid >> 5;
  const int half = lane >> 4, ml = lane & 15;
  const int j0 = blockIdx.x * 64, i0 = blockIdx.y * 64;
  const int bh = blockIdx.z;
  const int b = bh >> 4, h = bh & 15, kv = h >> 2;
  float* out = attn + (size_t)bh * SEQ * SEQ;

  if (j0 >= i0 + 64) {            // fully above diagonal: mask only
#pragma unroll
    for (int q = 0; q < 32; ++q) {
      int e = q * 128 + tid;
      int r = e >> 6, c = e & 63;
      out[(size_t)(i0 + r) * SEQ + (j0 + c)] = MASK_VAL;
    }
    return;
  }

  const __bf16* Q  = qbf + (size_t)(b * NH + h)   * SEQ * HD;
  const __bf16* Kp = kbf + (size_t)(b * NKV + kv) * SEQ * HD;
  // 64 rows x 128 B each = 512 x 16 B chunks per matrix; 4 chunks/thread/matrix
#pragma unroll
  for (int q = 0; q < 4; ++q) {
    int e = q * 128 + tid;
    int r = e >> 3, c = e & 7;                 // r = row, c = 16 B chunk (8 bf16)
    async_copy_b128(&sQ[r][c * 8], Q  + (size_t)(i0 + r) * HD + c * 8);
    async_copy_b128(&sK[r][c * 8], Kp + (size_t)(j0 + r) * HD + c * 8);
  }
  wait_async0();
  __syncthreads();

  const int wm = (wave >> 1) * 32, wn = (wave & 1) * 32;
  v8f acc[2][2] = {};
#pragma unroll
  for (int step = 0; step < 2; ++step) {
    v16bf a[2], bb[2];
#pragma unroll
    for (int t = 0; t < 2; ++t) {
      a[t]  = frag_row(&sQ[wm + t * 16 + ml][step * 32], half);
      bb[t] = frag_row(&sK[wn + t * 16 + ml][step * 32], half);
    }
#pragma unroll
    for (int tm = 0; tm < 2; ++tm)
#pragma unroll
      for (int tn = 0; tn < 2; ++tn)
        acc[tm][tn] = wmma_bf16(a[tm], bb[tn], acc[tm][tn]);
  }

#pragma unroll
  for (int tm = 0; tm < 2; ++tm)
#pragma unroll
    for (int tn = 0; tn < 2; ++tn)
#pragma unroll
      for (int r = 0; r < 8; ++r) {
        int i = i0 + wm + tm * 16 + r + 8 * half;
        int j = j0 + wn + tn * 16 + ml;
        float v = acc[tm][tn][r] * ATTN_SCALE;
        if (j > i) v = MASK_VAL;
        out[(size_t)i * SEQ + j] = v;
      }
}

// ---------------------------------------------------------------- softmax (wave per row)
__global__ void __launch_bounds__(256) softmax_kernel(float* __restrict__ attn) {
  const int lane = threadIdx.x & 31;
  const size_t row = (size_t)blockIdx.x * 8 + (threadIdx.x >> 5);
  float* p = attn + row * SEQ;
  float v[16];
  float m = MASK_VAL;
#pragma unroll
  for (int t = 0; t < 16; ++t) { v[t] = p[lane + t * 32]; m = fmaxf(m, v[t]); }
#pragma unroll
  for (int o = 16; o > 0; o >>= 1) m = fmaxf(m, __shfl_xor(m, o, 32));
  float s = 0.f;
#pragma unroll
  for (int t = 0; t < 16; ++t) { v[t] = __expf(v[t] - m); s += v[t]; }
#pragma unroll
  for (int o = 16; o > 0; o >>= 1) s += __shfl_xor(s, o, 32);
  float inv = 1.0f / s;
#pragma unroll
  for (int t = 0; t < 16; ++t) p[lane + t * 32] = v[t] * inv;
}

// ---------------------------------------------------------------- out = attn . sv
// per block: fixed (i, kv). O[16(b,r) x 64(d)] = A[16 x 512] * SV_i[512 x 64]
// 4 waves, each owns one 16-wide d slice; 16 K-steps of 32.
// SV staged TRANSPOSED ([d][j]) so the B fragment loads are contiguous b128s.
__global__ void __launch_bounds__(128)
outproj_kernel(const float* __restrict__ attn, const __bf16* __restrict__ svbf,
               float* __restrict__ headout) {
  __shared__ alignas(16) __bf16 sA[16][40];    // [rr][j]
  __shared__ alignas(16) __bf16 sVt[64][40];   // [d][j]
  const int i  = blockIdx.x;
  const int kv = blockIdx.y;
  const int tid  = threadIdx.x;
  const int lane = tid & 31, wave = tid >> 5;
  const int half = lane >> 4, ml = lane & 15;
  const int nb = wave * 16;

  v8f acc = {};
  for (int j0 = 0; j0 < SEQ; j0 += 32) {
#pragma unroll
    for (int q = 0; q < 4; ++q) {             // A: 16x32 from attn (f32->bf16)
      int e = q * 128 + tid;
      int rr = e >> 5, jj = e & 31;
      int b = rr >> 2, r = rr & 3;
      sA[rr][jj] = (__bf16)attn[((size_t)(b * NH + kv * 4 + r) * SEQ + i) * SEQ + j0 + jj];
    }
#pragma unroll
    for (int q = 0; q < 16; ++q) {            // SV rows sv[i][j0+jj][kv][:] -> sVt[d][jj]
      int e = q * 128 + tid;
      int jj = e >> 6, d = e & 63;            // coalesced global read along d
      sVt[d][jj] = svbf[(((size_t)i * SEQ + (j0 + jj)) * NKV + kv) * HD + d];
    }
    __syncthreads();
    v16bf a  = frag_row(&sA[ml][0], half);
    v16bf bb = frag_row(&sVt[nb + ml][0], half);
    acc = wmma_bf16(a, bb, acc);
    __syncthreads();
  }
#pragma unroll
  for (int r = 0; r < 8; ++r) {
    int rr = r + 8 * half;
    int b = rr >> 2, rloc = rr & 3;
    headout[((size_t)b * SEQ + i) * (NH * HD) + (kv * 4 + rloc) * HD + nb + ml] = acc[r];
  }
}

// ---------------------------------------------------------------- host
extern "C" void kernel_launch(void* const* d_in, const int* in_sizes, int n_in,
                              void* d_out, int out_size, void* d_ws, size_t ws_size,
                              hipStream_t stream) {
  (void)in_sizes; (void)n_in; (void)out_size; (void)ws_size;
  const float* x       = (const float*)d_in[0];
  const float* symbols = (const float*)d_in[1];
  const float* fcos    = (const float*)d_in[2];
  const float* fsin    = (const float*)d_in[3];
  const float* wq      = (const float*)d_in[4];
  const float* wk      = (const float*)d_in[5];
  const float* wv      = (const float*)d_in[6];
  const float* wo      = (const float*)d_in[7];
  float* out  = (float*)d_out;
  float* attn = out + (size_t)BSZ * SEQ * DM;

  char* ws = (char*)d_ws;
  size_t off = 0;
  auto take = [&](size_t bytes) -> char* {
    char* p = ws + off;
    off += (bytes + 255) & ~(size_t)255;
    return p;
  };
  __bf16* wq_t = (__bf16*)take((size_t)DM * DM * 2);     // [N=1024][K=1024]
  __bf16* wk_t = (__bf16*)take((size_t)DM * 256 * 2);    // [N=256][K=1024]
  __bf16* wv_t = (__bf16*)take((size_t)DM * 256 * 2);    // [N=256][K=1024]
  __bf16* wo_t = (__bf16*)take((size_t)DM * DM * 2);     // [N=1024][K=1024]
  float*  xq_tmp = (float*)take((size_t)BSZ * SEQ * DM * 4);
  float*  xk_tmp = (float*)take((size_t)BSZ * SEQ * 256 * 4);
  __bf16* qbf = (__bf16*)take((size_t)BSZ * NH * SEQ * HD * 2);
  __bf16* kbf = (__bf16*)take((size_t)BSZ * NKV * SEQ * HD * 2);
  __bf16* svbf = (__bf16*)take((size_t)SEQ * SEQ * NKV * HD * 2);   // 128 MiB
  float*  headout = (float*)take((size_t)BSZ * SEQ * NH * HD * 4);

  // 1. weight conversion f32 -> bf16, transposed to [N][K]
  cvt_transpose_bf16<<<(DM * DM + 255) / 256, 256, 0, stream>>>(wq, wq_t, DM, DM);
  cvt_transpose_bf16<<<(DM * 256 + 255) / 256, 256, 0, stream>>>(wk, wk_t, DM, 256);
  cvt_transpose_bf16<<<(DM * 256 + 255) / 256, 256, 0, stream>>>(wv, wv_t, DM, 256);
  cvt_transpose_bf16<<<(DM * DM + 255) / 256, 256, 0, stream>>>(wo, wo_t, DM, DM);

  // 2. projections (WMMA GEMMs, async-DMA B tiles)
  gemm_a32_b16<float><<<dim3(DM / 64, (BSZ * SEQ) / 64), 128, 0, stream>>>(
      x, wq_t, xq_tmp, BSZ * SEQ, DM, DM);
  gemm_a32_b16<float><<<dim3(256 / 64, (BSZ * SEQ) / 64), 128, 0, stream>>>(
      x, wk_t, xk_tmp, BSZ * SEQ, 256, DM);
  gemm_a32_b16<__bf16><<<dim3(256 / 64, (SEQ * SEQ) / 64), 128, 0, stream>>>(
      symbols, wv_t, svbf, SEQ * SEQ, 256, DM);     // dominant: 137 GF, 1 GB read

  // 3. RoPE -> head-major bf16
  rope_kernel<<<(BSZ * SEQ * NH * 32 + 255) / 256, 256, 0, stream>>>(xq_tmp, fcos, fsin, qbf, NH);
  rope_kernel<<<(BSZ * SEQ * NKV * 32 + 255) / 256, 256, 0, stream>>>(xk_tmp, fcos, fsin, kbf, NKV);

  // 4. scores + mask -> attn logits in d_out
  scores_kernel<<<dim3(SEQ / 64, SEQ / 64, BSZ * NH), 128, 0, stream>>>(qbf, kbf, attn);

  // 5. softmax in place
  softmax_kernel<<<(BSZ * NH * SEQ) / 8, 256, 0, stream>>>(attn);

  // 6. attn . sv -> per-head outputs
  outproj_kernel<<<dim3(SEQ, NKV), 128, 0, stream>>>(attn, svbf, headout);

  // 7. final projection
  gemm_a32_b16<float><<<dim3(DM / 64, (BSZ * SEQ) / 64), 128, 0, stream>>>(
      headout, wo_t, out, BSZ * SEQ, DM, DM);
}
